// SelfAttention_79783312490849
// MI455X (gfx1250) — matrix operand order repaired and checked by reading, not compiled, not run on hardware
//
#include <hip/hip_runtime.h>
#include <hip/hip_bf16.h>

typedef __attribute__((ext_vector_type(16))) _Float16 v16h;
typedef __attribute__((ext_vector_type(8)))  _Float16 v8h;
typedef __attribute__((ext_vector_type(8)))  float    v8f;

#define D_MODEL 1024
#define NHEAD   16
#define HD      64
#define SEQ     2048
#define BATCH   2
#define MTOT    (BATCH*SEQ)   // 4096
#define BH      (BATCH*NHEAD) // 32

#define WMMA_F32_F16(a,b,c) \
  __builtin_amdgcn_wmma_f32_16x16x32_f16(false,(a),false,(b),(short)0,(c),false,false)

// ---- Fragment loaders (layouts from CDNA5 ISA 7.12.2, wave32) ----
// A (16x32 f16, MxK): lane m=lane&15; elems 0..7 <-> K = koff..koff+7,
// elems 8..15 <-> K = 16+koff.. ; koff = 8*(lane>=16). Two 16B loads.
__device__ __forceinline__ v16h load_a16x32(const _Float16* __restrict__ base,
                                            int ldm, int lane) {
  const int row  = lane & 15;
  const int koff = (lane >> 4) << 3;
  const _Float16* p = base + row * ldm + koff;
  v8h lo = *(const v8h*)(p);
  v8h hi = *(const v8h*)(p + 16);
  v16h a;
#pragma unroll
  for (int i = 0; i < 8; ++i) { a[i] = lo[i]; a[i + 8] = hi[i]; }
  return a;
}

// B (32x16 f16, KxN) from K-contiguous ("transposed") storage [n][k]:
// lane n=lane&15; elems 0..7 <-> K = koff..koff+7, elems 8..15 <-> K = 16+koff..; koff = 8*(lane>=16). Two 16B loads.
__device__ __forceinline__ v16h load_b32x16(const _Float16* __restrict__ base,
                                            int ldn, int lane) {
  const int n    = lane & 15;
  const int koff = (lane >> 4) << 3;
  const _Float16* p = base + n * ldn + koff;
  v8h lo = *(const v8h*)(p);
  v8h hi = *(const v8h*)(p + 16);
  v16h b;
#pragma unroll
  for (int i = 0; i < 8; ++i) { b[i] = lo[i]; b[i + 8] = hi[i]; }
  return b;
}

// ---- Prep kernels ----
__global__ void cvt_f16_kernel(const float* __restrict__ in,
                               _Float16* __restrict__ out, int n) {
  int i = blockIdx.x * blockDim.x + threadIdx.x;
  if (i < n) out[i] = (_Float16)in[i];
}

// W: [K=1024][N=1024] f32 row-major  ->  Wt: [N][K] f16 (K-contiguous)
__global__ void transpose_f16_kernel(const float* __restrict__ W,
                                     _Float16* __restrict__ Wt) {
  int i = blockIdx.x * blockDim.x + threadIdx.x;
  int n = i >> 10;
  int k = i & 1023;
  Wt[i] = (_Float16)W[(k << 10) + n];
}

// ---- QKV projection GEMM: Y = X*W + b, scatter into attention layouts ----
// 32x64 tile per wave; K-loop unrolled x2 with ping-pong fragment sets
// (no cross-iteration register rotation -> no mov/NOP overhead).
// vmode==0: out[bh][s][hd] (Q,K)   vmode==1: out[bh][hd][s] (V transposed)
__global__ __launch_bounds__(32)
void gemm_qkv_kernel(const _Float16* __restrict__ X,
                     const _Float16* __restrict__ Wt,
                     const float* __restrict__ bias,
                     _Float16* __restrict__ out, int vmode) {
  const int lane = threadIdx.x;
  const int n0 = blockIdx.x * 64;
  const int m0 = blockIdx.y * 32;

  const _Float16* A0 = X + (size_t)m0 * D_MODEL;
  const _Float16* A1 = X + (size_t)(m0 + 16) * D_MODEL;
  const _Float16* B0 = Wt + (size_t)(n0 +  0) * D_MODEL;
  const _Float16* B1 = Wt + (size_t)(n0 + 16) * D_MODEL;
  const _Float16* B2 = Wt + (size_t)(n0 + 32) * D_MODEL;
  const _Float16* B3 = Wt + (size_t)(n0 + 48) * D_MODEL;

  // set A: fragments for k0 ; set B: fragments for k0+32
  v16h a0A = load_a16x32(A0, D_MODEL, lane);
  v16h a1A = load_a16x32(A1, D_MODEL, lane);
  v16h b0A = load_b32x16(B0, D_MODEL, lane);
  v16h b1A = load_b32x16(B1, D_MODEL, lane);
  v16h b2A = load_b32x16(B2, D_MODEL, lane);
  v16h b3A = load_b32x16(B3, D_MODEL, lane);
  v16h a0B, a1B, b0B, b1B, b2B, b3B;

  v8f acc[2][4] = {};
  for (int k0 = 0; k0 < D_MODEL; k0 += 64) {
    const int k1 = k0 + 32;
    const int k2 = k0 + 64;
    // issue set-B loads (k0+32), then consume set A
    a0B = load_a16x32(A0 + k1, D_MODEL, lane);
    a1B = load_a16x32(A1 + k1, D_MODEL, lane);
    b0B = load_b32x16(B0 + k1, D_MODEL, lane);
    b1B = load_b32x16(B1 + k1, D_MODEL, lane);
    b2B = load_b32x16(B2 + k1, D_MODEL, lane);
    b3B = load_b32x16(B3 + k1, D_MODEL, lane);
    __builtin_prefetch(A0 + k1 + 32, 0, 3);
    acc[0][0] = WMMA_F32_F16(a0A, b0A, acc[0][0]);
    acc[0][1] = WMMA_F32_F16(a0A, b1A, acc[0][1]);
    acc[0][2] = WMMA_F32_F16(a0A, b2A, acc[0][2]);
    acc[0][3] = WMMA_F32_F16(a0A, b3A, acc[0][3]);
    acc[1][0] = WMMA_F32_F16(a1A, b0A, acc[1][0]);
    acc[1][1] = WMMA_F32_F16(a1A, b1A, acc[1][1]);
    acc[1][2] = WMMA_F32_F16(a1A, b2A, acc[1][2]);
    acc[1][3] = WMMA_F32_F16(a1A, b3A, acc[1][3]);
    // issue set-A loads (k0+64), then consume set B
    if (k2 < D_MODEL) {
      a0A = load_a16x32(A0 + k2, D_MODEL, lane);
      a1A = load_a16x32(A1 + k2, D_MODEL, lane);
      b0A = load_b32x16(B0 + k2, D_MODEL, lane);
      b1A = load_b32x16(B1 + k2, D_MODEL, lane);
      b2A = load_b32x16(B2 + k2, D_MODEL, lane);
      b3A = load_b32x16(B3 + k2, D_MODEL, lane);
    }
    acc[0][0] = WMMA_F32_F16(a0B, b0B, acc[0][0]);
    acc[0][1] = WMMA_F32_F16(a0B, b1B, acc[0][1]);
    acc[0][2] = WMMA_F32_F16(a0B, b2B, acc[0][2]);
    acc[0][3] = WMMA_F32_F16(a0B, b3B, acc[0][3]);
    acc[1][0] = WMMA_F32_F16(a1B, b0B, acc[1][0]);
    acc[1][1] = WMMA_F32_F16(a1B, b1B, acc[1][1]);
    acc[1][2] = WMMA_F32_F16(a1B, b2B, acc[1][2]);
    acc[1][3] = WMMA_F32_F16(a1B, b3B, acc[1][3]);
  }

  const int hi = lane >> 4;
  const int nl = lane & 15;
  const int bb = m0 >> 11;          // batch (32-row tiles never cross S=2048)
#pragma unroll
  for (int mb = 0; mb < 2; ++mb)
#pragma unroll
    for (int nb = 0; nb < 4; ++nb) {
      const int n = n0 + nb * 16 + nl;
      const float bn = bias[n];
      const int h = n >> 6, d = n & 63;
#pragma unroll
      for (int r = 0; r < 8; ++r) {
        const int m = m0 + mb * 16 + r + 8 * hi;
        const int s = m & (SEQ - 1);
        const float v = acc[mb][nb][r] + bn;
        size_t idx;
        if (vmode) idx = ((size_t)(bb * NHEAD + h) * HD + d) * SEQ + s;
        else       idx = ((size_t)(bb * NHEAD + h) * SEQ + s) * HD + d;
        out[idx] = (_Float16)v;
      }
    }
}

// ---- Flash attention: one wave per (bh, 32-query block) ----
// 16 WMMAs per 32-key step; K^T fragments ping-pong between two sets
// (j-loop unrolled x2), V loads hoisted above the softmax barriers.
__global__ __launch_bounds__(32)
void attention_kernel(const _Float16* __restrict__ qh,  // [BH][S][64]
                      const _Float16* __restrict__ kh,  // [BH][S][64]
                      const _Float16* __restrict__ vt,  // [BH][64][S]
                      _Float16* __restrict__ ctx) {     // [B][S][1024] f16
  __shared__ __align__(16) float    sS[32 * 32];
  __shared__ __align__(16) _Float16 sP[32 * 32];
  __shared__ float sAlpha[32];
  __shared__ float sInvL[32];

  const int lane = threadIdx.x;
  const int m0 = blockIdx.x * 32;
  const int bh = blockIdx.y;
  const int bb = bh >> 4;
  const int h  = bh & 15;
  const int hi = lane >> 4;
  const int nl = lane & 15;

  const _Float16* qbase = qh + (size_t)bh * SEQ * HD;
  const _Float16* kbase = kh + (size_t)bh * SEQ * HD;
  const _Float16* vbase = vt + (size_t)bh * HD * SEQ;

  v16h aQ[2][2];
#pragma unroll
  for (int mb = 0; mb < 2; ++mb) {
    aQ[mb][0] = load_a16x32(qbase + (m0 + mb * 16) * HD + 0,  HD, lane);
    aQ[mb][1] = load_a16x32(qbase + (m0 + mb * 16) * HD + 32, HD, lane);
  }

  // Two ping-pong sets of K^T B-fragments: [key-sub-block][kc]
  v16h bKa[2][2], bKb[2][2];
  bKa[0][0] = load_b32x16(kbase + (size_t)0  * HD + 0,  HD, lane);
  bKa[0][1] = load_b32x16(kbase + (size_t)0  * HD + 32, HD, lane);
  bKa[1][0] = load_b32x16(kbase + (size_t)16 * HD + 0,  HD, lane);
  bKa[1][1] = load_b32x16(kbase + (size_t)16 * HD + 32, HD, lane);

  v8f o[2][4] = {};
  float mrow = -3.0e38f, lrow = 0.0f;   // each lane owns query row m0+lane

  // one 32-key step: consume bKc (keys at j), preload bKn (keys at jn)
  auto step = [&](v16h (&bKc)[2][2], v16h (&bKn)[2][2], int j, int jn) {
    // ---- scores: 32x32 tile = Q(32x64) * K^T ----
    v8f s[2][2] = {};
#pragma unroll
    for (int mb = 0; mb < 2; ++mb)
#pragma unroll
      for (int kb = 0; kb < 2; ++kb) {
        s[mb][kb] = WMMA_F32_F16(aQ[mb][0], bKc[kb][0], s[mb][kb]);
        s[mb][kb] = WMMA_F32_F16(aQ[mb][1], bKc[kb][1], s[mb][kb]);
      }

    // V B-fragments for this step (independent of softmax -> issue early)
    v16h bV[4];
#pragma unroll
    for (int nb = 0; nb < 4; ++nb)
      bV[nb] = load_b32x16(vbase + (size_t)(nb * 16) * SEQ + j, SEQ, lane);

    // next-step K fragments (overlap the softmax below)
    if (jn < SEQ) {
      bKn[0][0] = load_b32x16(kbase + (size_t)(jn + 0)  * HD + 0,  HD, lane);
      bKn[0][1] = load_b32x16(kbase + (size_t)(jn + 0)  * HD + 32, HD, lane);
      bKn[1][0] = load_b32x16(kbase + (size_t)(jn + 16) * HD + 0,  HD, lane);
      bKn[1][1] = load_b32x16(kbase + (size_t)(jn + 16) * HD + 32, HD, lane);
    }

#pragma unroll
    for (int mb = 0; mb < 2; ++mb)
#pragma unroll
      for (int kb = 0; kb < 2; ++kb)
#pragma unroll
        for (int r = 0; r < 8; ++r)
          sS[(mb * 16 + r + 8 * hi) * 32 + kb * 16 + nl] = s[mb][kb][r] * 0.125f;
    __syncthreads();

    // ---- online softmax: each lane owns one of the 32 query rows ----
    {
      float bm = -3.0e38f;
#pragma unroll
      for (int c = 0; c < 32; ++c) bm = fmaxf(bm, sS[lane * 32 + c]);
      const float mnew  = fmaxf(mrow, bm);
      const float alpha = __expf(mrow - mnew);
      float ls = 0.0f;
#pragma unroll
      for (int c = 0; c < 32; ++c) {
        const float p = __expf(sS[lane * 32 + c] - mnew);
        sP[lane * 32 + c] = (_Float16)p;
        ls += p;
      }
      lrow = lrow * alpha + ls;
      mrow = mnew;
      sAlpha[lane] = alpha;
    }
    __syncthreads();

#pragma unroll
    for (int mb = 0; mb < 2; ++mb) {
      float arow[8];
#pragma unroll
      for (int r = 0; r < 8; ++r) arow[r] = sAlpha[mb * 16 + r + 8 * hi];
#pragma unroll
      for (int nb = 0; nb < 4; ++nb)
#pragma unroll
        for (int r = 0; r < 8; ++r) o[mb][nb][r] *= arow[r];
    }

    v16h aP0 = load_a16x32(sP,           32, lane);
    v16h aP1 = load_a16x32(sP + 16 * 32, 32, lane);
#pragma unroll
    for (int nb = 0; nb < 4; ++nb) {
      o[0][nb] = WMMA_F32_F16(aP0, bV[nb], o[0][nb]);
      o[1][nb] = WMMA_F32_F16(aP1, bV[nb], o[1][nb]);
    }
    __syncthreads();   // protect sS/sP before next step overwrites
  };

  for (int j = 0; j < SEQ; j += 64) {
    step(bKa, bKb, j,      j + 32);
    step(bKb, bKa, j + 32, j + 64);
  }

  sInvL[lane] = 1.0f / lrow;
  __syncthreads();
#pragma unroll
  for (int mb = 0; mb < 2; ++mb) {
    float invl[8];
#pragma unroll
    for (int r = 0; r < 8; ++r) invl[r] = sInvL[mb * 16 + r + 8 * hi];
#pragma unroll
    for (int nb = 0; nb < 4; ++nb)
#pragma unroll
      for (int r = 0; r < 8; ++r) {
        const int m = m0 + mb * 16 + r + 8 * hi;
        const size_t idx = ((size_t)bb * SEQ + m) * D_MODEL + h * HD + nb * 16 + nl;
        ctx[idx] = (_Float16)(o[mb][nb][r] * invl[r]);
      }
  }
}

// ---- Output projection: Y = relu(ctx*Wo + bo), f32 out ----
__global__ __launch_bounds__(32)
void gemm_out_kernel(const _Float16* __restrict__ X,   // [MTOT][1024]
                     const _Float16* __restrict__ Wt,  // [1024][1024]
                     const float* __restrict__ bias,
                     float* __restrict__ Y) {
  const int lane = threadIdx.x;
  const int n0 = blockIdx.x * 64;
  const int m0 = blockIdx.y * 32;

  const _Float16* A0 = X + (size_t)m0 * D_MODEL;
  const _Float16* A1 = X + (size_t)(m0 + 16) * D_MODEL;
  const _Float16* B0 = Wt + (size_t)(n0 +  0) * D_MODEL;
  const _Float16* B1 = Wt + (size_t)(n0 + 16) * D_MODEL;
  const _Float16* B2 = Wt + (size_t)(n0 + 32) * D_MODEL;
  const _Float16* B3 = Wt + (size_t)(n0 + 48) * D_MODEL;

  v16h a0A = load_a16x32(A0, D_MODEL, lane);
  v16h a1A = load_a16x32(A1, D_MODEL, lane);
  v16h b0A = load_b32x16(B0, D_MODEL, lane);
  v16h b1A = load_b32x16(B1, D_MODEL, lane);
  v16h b2A = load_b32x16(B2, D_MODEL, lane);
  v16h b3A = load_b32x16(B3, D_MODEL, lane);
  v16h a0B, a1B, b0B, b1B, b2B, b3B;

  v8f acc[2][4] = {};
  for (int k0 = 0; k0 < D_MODEL; k0 += 64) {
    const int k1 = k0 + 32;
    const int k2 = k0 + 64;
    a0B = load_a16x32(A0 + k1, D_MODEL, lane);
    a1B = load_a16x32(A1 + k1, D_MODEL, lane);
    b0B = load_b32x16(B0 + k1, D_MODEL, lane);
    b1B = load_b32x16(B1 + k1, D_MODEL, lane);
    b2B = load_b32x16(B2 + k1, D_MODEL, lane);
    b3B = load_b32x16(B3 + k1, D_MODEL, lane);
    __builtin_prefetch(A0 + k1 + 32, 0, 3);
    acc[0][0] = WMMA_F32_F16(a0A, b0A, acc[0][0]);
    acc[0][1] = WMMA_F32_F16(a0A, b1A, acc[0][1]);
    acc[0][2] = WMMA_F32_F16(a0A, b2A, acc[0][2]);
    acc[0][3] = WMMA_F32_F16(a0A, b3A, acc[0][3]);
    acc[1][0] = WMMA_F32_F16(a1A, b0A, acc[1][0]);
    acc[1][1] = WMMA_F32_F16(a1A, b1A, acc[1][1]);
    acc[1][2] = WMMA_F32_F16(a1A, b2A, acc[1][2]);
    acc[1][3] = WMMA_F32_F16(a1A, b3A, acc[1][3]);
    if (k2 < D_MODEL) {
      a0A = load_a16x32(A0 + k2, D_MODEL, lane);
      a1A = load_a16x32(A1 + k2, D_MODEL, lane);
      b0A = load_b32x16(B0 + k2, D_MODEL, lane);
      b1A = load_b32x16(B1 + k2, D_MODEL, lane);
      b2A = load_b32x16(B2 + k2, D_MODEL, lane);
      b3A = load_b32x16(B3 + k2, D_MODEL, lane);
    }
    acc[0][0] = WMMA_F32_F16(a0B, b0B, acc[0][0]);
    acc[0][1] = WMMA_F32_F16(a0B, b1B, acc[0][1]);
    acc[0][2] = WMMA_F32_F16(a0B, b2B, acc[0][2]);
    acc[0][3] = WMMA_F32_F16(a0B, b3B, acc[0][3]);
    acc[1][0] = WMMA_F32_F16(a1B, b0B, acc[1][0]);
    acc[1][1] = WMMA_F32_F16(a1B, b1B, acc[1][1]);
    acc[1][2] = WMMA_F32_F16(a1B, b2B, acc[1][2]);
    acc[1][3] = WMMA_F32_F16(a1B, b3B, acc[1][3]);
  }

  const int hi = lane >> 4;
  const int nl = lane & 15;
#pragma unroll
  for (int mb = 0; mb < 2; ++mb)
#pragma unroll
    for (int nb = 0; nb < 4; ++nb) {
      const int n = n0 + nb * 16 + nl;
      const float bn = bias[n];
#pragma unroll
      for (int r = 0; r < 8; ++r) {
        const int m = m0 + mb * 16 + r + 8 * hi;
        Y[(size_t)m * D_MODEL + n] = fmaxf(acc[mb][nb][r] + bn, 0.0f);
      }
    }
}

// ---- LayerNorm over last dim (1024), one block per row ----
__global__ __launch_bounds__(256)
void layernorm_kernel(const float* __restrict__ Y, float* __restrict__ out) {
  __shared__ float red[256];
  __shared__ float red2[256];
  const int row = blockIdx.x;
  const float* y = Y + (size_t)row * D_MODEL;
  float s = 0.0f, s2 = 0.0f;
  for (int i = threadIdx.x; i < D_MODEL; i += 256) {
    const float v = y[i];
    s += v; s2 += v * v;
  }
  red[threadIdx.x] = s; red2[threadIdx.x] = s2;
  __syncthreads();
  for (int st = 128; st > 0; st >>= 1) {
    if (threadIdx.x < st) {
      red[threadIdx.x]  += red[threadIdx.x + st];
      red2[threadIdx.x] += red2[threadIdx.x + st];
    }
    __syncthreads();
  }
  const float mu   = red[0] * (1.0f / D_MODEL);
  const float var  = red2[0] * (1.0f / D_MODEL) - mu * mu;
  const float rstd = rsqrtf(var + 1e-5f);
  for (int i = threadIdx.x; i < D_MODEL; i += 256)
    out[(size_t)row * D_MODEL + i] = (y[i] - mu) * rstd;
}

extern "C" void kernel_launch(void* const* d_in, const int* in_sizes, int n_in,
                              void* d_out, int out_size, void* d_ws, size_t ws_size,
                              hipStream_t stream) {
  const float* x  = (const float*)d_in[0];
  const float* Wq = (const float*)d_in[1];
  const float* bq = (const float*)d_in[2];
  const float* Wk = (const float*)d_in[3];
  const float* bk = (const float*)d_in[4];
  const float* Wv = (const float*)d_in[5];
  const float* bv = (const float*)d_in[6];
  const float* Wo = (const float*)d_in[7];
  const float* bo = (const float*)d_in[8];
  float* out = (float*)d_out;

  char* ws = (char*)d_ws;
  const size_t SZ_XH = (size_t)MTOT * D_MODEL * 2;      // 8 MB
  const size_t SZ_W  = (size_t)D_MODEL * D_MODEL * 2;   // 2 MB each
  const size_t SZ_QK = (size_t)BH * SEQ * HD * 2;       // 8 MB each

  _Float16* xh  = (_Float16*)(ws);                       // [0, 8M) (aliased by ctx)
  _Float16* wqt = (_Float16*)(ws + SZ_XH);
  _Float16* wkt = (_Float16*)(ws + SZ_XH + SZ_W);
  _Float16* wvt = (_Float16*)(ws + SZ_XH + 2 * SZ_W);
  _Float16* wot = (_Float16*)(ws + SZ_XH + 3 * SZ_W);
  _Float16* qh  = (_Float16*)(ws + SZ_XH + 4 * SZ_W);            // 8 MB
  _Float16* kh  = (_Float16*)(ws + SZ_XH + 4 * SZ_W + SZ_QK);    // 8 MB
  _Float16* vt  = (_Float16*)(ws + SZ_XH + 4 * SZ_W + 2 * SZ_QK);// 8 MB
  _Float16* ctx = xh;              // xh dead after QKV GEMMs
  float*    Yb  = (float*)qh;      // q/k dead after attention (16 MB spans qh+kh)

  cvt_f16_kernel<<<(MTOT * D_MODEL) / 256, 256, 0, stream>>>(x, xh, MTOT * D_MODEL);
  const int tgrid = (D_MODEL * D_MODEL) / 256;
  transpose_f16_kernel<<<tgrid, 256, 0, stream>>>(Wq, wqt);
  transpose_f16_kernel<<<tgrid, 256, 0, stream>>>(Wk, wkt);
  transpose_f16_kernel<<<tgrid, 256, 0, stream>>>(Wv, wvt);
  transpose_f16_kernel<<<tgrid, 256, 0, stream>>>(Wo, wot);

  dim3 gg(D_MODEL / 64, MTOT / 32);
  gemm_qkv_kernel<<<gg, 32, 0, stream>>>(xh, wqt, bq, qh, 0);
  gemm_qkv_kernel<<<gg, 32, 0, stream>>>(xh, wkt, bk, kh, 0);
  gemm_qkv_kernel<<<gg, 32, 0, stream>>>(xh, wvt, bv, vt, 1);

  dim3 ga(SEQ / 32, BH);
  attention_kernel<<<ga, 32, 0, stream>>>(qh, kh, vt, ctx);

  gemm_out_kernel<<<gg, 32, 0, stream>>>(ctx, wot, bo, Yb);
  layernorm_kernel<<<MTOT, 256, 0, stream>>>(Yb, out);
}
